// HeteroGLSTM_57612691309064
// MI455X (gfx1250) — compile-verified, matrix operand order checked by reading, not run on hardware
//
#include <hip/hip_runtime.h>
#include <cstdint>
#include <cstddef>

// ---------------------------------------------------------------------------
// Types for WMMA
// ---------------------------------------------------------------------------
typedef __attribute__((ext_vector_type(16))) __bf16 v16bf;
typedef __attribute__((ext_vector_type(8)))  float  v8f;
typedef __attribute__((ext_vector_type(4)))  unsigned int u32x4;

union FragU { v16bf v; u32x4 q[2]; };

__device__ __forceinline__ unsigned short f2bf(float f) {
    unsigned int u = __float_as_uint(f);
    unsigned int r = u + 0x7fffu + ((u >> 16) & 1u);   // round-to-nearest-even
    return (unsigned short)(r >> 16);
}

__device__ __forceinline__ float sigf(float x) { return 1.0f / (1.0f + __expf(-x)); }

// Gate storage order (jax sorted keys): 0=c(tanh), 1=f, 2=i, 3=o
struct GateParams {
    const float* Wdst[4][3];   // [gate][edge-type e0,e1,e2]  128x256
    const float* Wsrc[4][3];   // 128x256
    const float* b[4][3];      // 256
    const float* linW[4][2];   // [gate][ntype n0,n1] 256x256
    const float* linb[4][2];   // 256
};

// ---------------------------------------------------------------------------
// Zero scratch region (f32)
// ---------------------------------------------------------------------------
__global__ void zero_f32(float* p, long long n) {
    long long i = (long long)blockIdx.x * blockDim.x + threadIdx.x;
    long long st = (long long)gridDim.x * blockDim.x;
    for (; i < n; i += st) p[i] = 0.0f;
}

// ---------------------------------------------------------------------------
// Scatter-sum: one wave per edge, each lane handles 4 of 128 features
// ---------------------------------------------------------------------------
__global__ void scatter_sum(const float* __restrict__ xsrc, const int* __restrict__ e,
                            long long E, float* __restrict__ sum, float* __restrict__ cnt) {
    int lane = threadIdx.x & 31;
    long long wid = ((long long)blockIdx.x * blockDim.x + threadIdx.x) >> 5;
    long long nw  = ((long long)gridDim.x * blockDim.x) >> 5;
    for (long long ei = wid; ei < E; ei += nw) {
        int src = e[ei];
        int dst = e[E + ei];
        const float4 v = *(const float4*)(xsrc + (long long)src * 128 + lane * 4);
        float* d = sum + (long long)dst * 128 + lane * 4;
        atomicAdd(d + 0, v.x);
        atomicAdd(d + 1, v.y);
        atomicAdd(d + 2, v.z);
        atomicAdd(d + 3, v.w);
        if (lane == 0) atomicAdd(cnt + dst, 1.0f);
    }
}

// ---------------------------------------------------------------------------
// Finalize means and pack bf16 activation matrices
// A0 = [mean0 | x0]  -> [N,256] ; A1 = [mean1_e0 | x1 | mean1_e2] -> [N,384]
// ---------------------------------------------------------------------------
__global__ void pack_a0(const float* __restrict__ sum0, const float* __restrict__ cnt0,
                        const float* __restrict__ x0, unsigned short* __restrict__ A0, int N) {
    long long i = (long long)blockIdx.x * blockDim.x + threadIdx.x;
    long long tot = (long long)N * 128, st = (long long)gridDim.x * blockDim.x;
    for (; i < tot; i += st) {
        int n = (int)(i >> 7), k = (int)(i & 127);
        float c = cnt0[n]; c = c < 1.0f ? 1.0f : c;
        A0[(long long)n * 256 + k]       = f2bf(sum0[i] / c);
        A0[(long long)n * 256 + 128 + k] = f2bf(x0[i]);
    }
}

__global__ void pack_a1(const float* __restrict__ sa, const float* __restrict__ ca,
                        const float* __restrict__ sc, const float* __restrict__ cc,
                        const float* __restrict__ x1, unsigned short* __restrict__ A1, int N) {
    long long i = (long long)blockIdx.x * blockDim.x + threadIdx.x;
    long long tot = (long long)N * 128, st = (long long)gridDim.x * blockDim.x;
    for (; i < tot; i += st) {
        int n = (int)(i >> 7), k = (int)(i & 127);
        float c1 = ca[n]; c1 = c1 < 1.0f ? 1.0f : c1;
        float c2 = cc[n]; c2 = c2 < 1.0f ? 1.0f : c2;
        A1[(long long)n * 384 + k]       = f2bf(sa[i] / c1);
        A1[(long long)n * 384 + 128 + k] = f2bf(x1[i]);
        A1[(long long)n * 384 + 256 + k] = f2bf(sc[i] / c2);
    }
}

// ---------------------------------------------------------------------------
// Pack transposed bf16 weights + fused biases.
//   W0T [1024 x 256]: out-chan n=g*256+j; k<128 -> Wsrc_e1[k][j], else Wdst_e1[k-128][j]
//   W1T [1024 x 384]: k<128 Wsrc_e0; k<256 Wdst_e0+Wdst_e2; else Wsrc_e2
//   L0T/L1T [1024 x 256]: row n holds lin_n?_W[:,j] for gate g
// ---------------------------------------------------------------------------
__global__ void pack_weights(GateParams P,
                             unsigned short* __restrict__ W0T, unsigned short* __restrict__ W1T,
                             unsigned short* __restrict__ L0T, unsigned short* __restrict__ L1T,
                             float* __restrict__ bias0, float* __restrict__ bias1,
                             float* __restrict__ lb0, float* __restrict__ lb1) {
    long long i = (long long)blockIdx.x * blockDim.x + threadIdx.x;
    long long tot = 1024LL * 384, st = (long long)gridDim.x * blockDim.x;
    for (; i < tot; i += st) {
        int n = (int)(i / 384), k = (int)(i % 384);
        int g = n >> 8, j = n & 255;
        // W1T (K=384)
        float w1;
        if (k < 128)       w1 = P.Wsrc[g][0][k * 256 + j];
        else if (k < 256)  w1 = P.Wdst[g][0][(k - 128) * 256 + j] + P.Wdst[g][2][(k - 128) * 256 + j];
        else               w1 = P.Wsrc[g][2][(k - 256) * 256 + j];
        W1T[(long long)n * 384 + k] = f2bf(w1);
        if (k < 256) {
            float w0 = (k < 128) ? P.Wsrc[g][1][k * 256 + j]
                                 : P.Wdst[g][1][(k - 128) * 256 + j];
            W0T[(long long)n * 256 + k] = f2bf(w0);
            L0T[(long long)n * 256 + k] = f2bf(P.linW[g][0][k * 256 + j]);
            L1T[(long long)n * 256 + k] = f2bf(P.linW[g][1][k * 256 + j]);
        }
        if (k == 0) {
            bias0[n] = P.b[g][1][j];
            bias1[n] = P.b[g][0][j] + P.b[g][2][j];
            lb0[n]   = P.linb[g][0][j];
            lb1[n]   = P.linb[g][1][j];
        }
    }
}

// ---------------------------------------------------------------------------
// WMMA GEMM: out[N x 1024] = A[N x K(slice)] * BT^T + bias
//  - B tile [64 x K] staged in LDS once per block (8x less L2 traffic),
//    rows padded by 8 halfwords so ds_load_b128 rotates across all 64 banks.
//  - Each wave owns a 32x64 tile: 2 A frags, 8 accumulators,
//    8 independent v_wmma_f32_16x16x32_bf16 per K-step.
//  - aColShift: per-gate column offset into A for the block-diagonal GEMM2.
// ---------------------------------------------------------------------------
#define GEMM_WAVES 8
#define MAXKP 392   // max K (384) + 8-halfword pad

template <int OUT_BF16>
__global__ void gemm_wmma(const unsigned short* __restrict__ A, int lda,
                          const unsigned short* __restrict__ BT, int K,
                          const float* __restrict__ bias,
                          void* __restrict__ outP, int ldo,
                          int nRowTiles32, int aColShift) {
    __shared__ unsigned short Bs[64 * MAXKP];
    const int KP = K + 8;
    const int lane = threadIdx.x & 31;
    const int wave = threadIdx.x >> 5;
    const int colBase = blockIdx.y * 64;

    // ---- cooperative stage of B tile [64 x K] into LDS ----
    {
        int chunks = (64 * K) >> 3;          // 16-byte chunks
        for (int idx = threadIdx.x; idx < chunks; idx += blockDim.x) {
            int el  = idx << 3;
            int row = el / K;
            int col = el - row * K;
            *(u32x4*)(&Bs[row * KP + col]) =
                *(const u32x4*)(BT + (size_t)(colBase + row) * K + col);
        }
    }
    __syncthreads();

    int rowTile = blockIdx.x * GEMM_WAVES + wave;
    if (rowTile >= nRowTiles32) return;

    const int aOff = (blockIdx.y >> 2) * aColShift;
    const int m    = lane & 15;
    const int half = lane >> 4;

    const unsigned short* aPtr0 = A + (size_t)(rowTile * 32 + m) * lda + aOff + half * 8;
    const unsigned short* aPtr1 = aPtr0 + (size_t)16 * lda;
    const unsigned short* bLds  = &Bs[m * KP + half * 8];

    v8f acc0[4] = {};
    v8f acc1[4] = {};
    for (int ks = 0; ks < K; ks += 32) {
        FragU a0, a1;
        a0.q[0] = *(const u32x4*)(aPtr0 + ks);
        a0.q[1] = *(const u32x4*)(aPtr0 + ks + 16);
        a1.q[0] = *(const u32x4*)(aPtr1 + ks);
        a1.q[1] = *(const u32x4*)(aPtr1 + ks + 16);
#pragma unroll
        for (int j = 0; j < 4; ++j) {
            FragU b;
            const unsigned short* bp = bLds + j * 16 * KP + ks;
            b.q[0] = *(const u32x4*)(bp);
            b.q[1] = *(const u32x4*)(bp + 16);
            acc0[j] = __builtin_amdgcn_wmma_f32_16x16x32_bf16(
                false, a0.v, false, b.v, (short)0, acc0[j], false, false);
            acc1[j] = __builtin_amdgcn_wmma_f32_16x16x32_bf16(
                false, a1.v, false, b.v, (short)0, acc1[j], false, false);
        }
    }

    const int orow0 = rowTile * 32 + half * 8;
#pragma unroll
    for (int j = 0; j < 4; ++j) {
        int ocol = colBase + j * 16 + m;
        float bb = bias[ocol];
#pragma unroll
        for (int i = 0; i < 8; ++i) {
            float v0 = acc0[j][i] + bb;
            float v1 = acc1[j][i] + bb;
            if (OUT_BF16) {
                ((unsigned short*)outP)[(size_t)(orow0 + i) * ldo + ocol]      = f2bf(v0);
                ((unsigned short*)outP)[(size_t)(orow0 + 16 + i) * ldo + ocol] = f2bf(v1);
            } else {
                ((float*)outP)[(size_t)(orow0 + i) * ldo + ocol]      = v0;
                ((float*)outP)[(size_t)(orow0 + 16 + i) * ldo + ocol] = v1;
            }
        }
    }
}

// ---------------------------------------------------------------------------
// Activations + LSTM combine. Gate storage order: [c(tanh), f, i, o].
//   c_new = sig(f)*c_prev + sig(i)*tanh(c_raw) ;  h = sig(o)*tanh(c_new)
// ---------------------------------------------------------------------------
__global__ void lstm_combine(const float* __restrict__ Y, const float* __restrict__ cprev,
                             float* __restrict__ h_out, float* __restrict__ c_out, int N) {
    long long i = (long long)blockIdx.x * blockDim.x + threadIdx.x;
    long long tot = (long long)N * 256, st = (long long)gridDim.x * blockDim.x;
    for (; i < tot; i += st) {
        int n = (int)(i >> 8), ch = (int)(i & 255);
        const float* y = Y + (size_t)n * 1024 + ch;
        float tg = tanhf(y[0]);
        float fg = sigf(y[256]);
        float ig = sigf(y[512]);
        float og = sigf(y[768]);
        float cn = fg * cprev[i] + ig * tg;
        c_out[i] = cn;
        h_out[i] = og * tanhf(cn);
    }
}

// ---------------------------------------------------------------------------
// Host launcher
// ---------------------------------------------------------------------------
extern "C" void kernel_launch(void* const* d_in, const int* in_sizes, int n_in,
                              void* d_out, int out_size, void* d_ws, size_t ws_size,
                              hipStream_t stream) {
    const int D = 128, HH = 256;
    const int N0 = in_sizes[0] / D;
    const int N1 = in_sizes[1] / D;
    const long long E0 = in_sizes[4] / 2, E1 = in_sizes[5] / 2, E2 = in_sizes[6] / 2;

    const float* x0 = (const float*)d_in[0];
    const float* x1 = (const float*)d_in[1];
    const float* c0 = (const float*)d_in[2];
    const float* c1 = (const float*)d_in[3];
    const int* e0 = (const int*)d_in[4];
    const int* e1 = (const int*)d_in[5];
    const int* e2 = (const int*)d_in[6];

    // ---- param leaves: either 52 separate buffers or one flat buffer ----
    const float* leaves[52];
    static const int leaf_sz[13] = {128*256, 128*256, 256,   // e0: W_dst, W_src, b
                                    128*256, 128*256, 256,   // e1
                                    128*256, 128*256, 256,   // e2
                                    256*256, 256,            // lin_n0_W, lin_n0_b
                                    256*256, 256};           // lin_n1_W, lin_n1_b
    if (n_in >= 7 + 52) {
        for (int i = 0; i < 52; ++i) leaves[i] = (const float*)d_in[7 + i];
    } else {
        const float* p = (const float*)d_in[7];
        size_t off = 0; int li = 0;
        for (int g = 0; g < 4; ++g)
            for (int j = 0; j < 13; ++j) { leaves[li++] = p + off; off += leaf_sz[j]; }
    }
    GateParams P;
    for (int g = 0; g < 4; ++g) {
        for (int et = 0; et < 3; ++et) {
            P.Wdst[g][et] = leaves[g * 13 + et * 3 + 0];
            P.Wsrc[g][et] = leaves[g * 13 + et * 3 + 1];
            P.b[g][et]    = leaves[g * 13 + et * 3 + 2];
        }
        P.linW[g][0] = leaves[g * 13 + 9];
        P.linb[g][0] = leaves[g * 13 + 10];
        P.linW[g][1] = leaves[g * 13 + 11];
        P.linb[g][1] = leaves[g * 13 + 12];
    }

    // ---- workspace layout ----
    size_t off = 0;
    auto alloc = [&](size_t bytes) {
        size_t o = off;
        off = (off + bytes + 255) & ~(size_t)255;
        return o;
    };
    char* ws = (char*)d_ws;
    size_t o_sum0  = alloc((size_t)N0 * D * 4);
    size_t o_cnt0  = alloc((size_t)N0 * 4);
    size_t o_sum1a = alloc((size_t)N1 * D * 4);
    size_t o_cnt1a = alloc((size_t)N1 * 4);
    size_t o_sum1c = alloc((size_t)N1 * D * 4);
    size_t o_cnt1c = alloc((size_t)N1 * 4);
    size_t zeroEnd = off;                         // contiguous region to clear
    size_t o_A0  = alloc((size_t)N0 * 256 * 2);
    size_t o_A1  = alloc((size_t)N1 * 384 * 2);
    size_t o_W0T = alloc(1024 * 256 * 2);
    size_t o_W1T = alloc(1024 * 384 * 2);
    size_t o_L0T = alloc(1024 * 256 * 2);
    size_t o_L1T = alloc(1024 * 256 * 2);
    size_t o_b0  = alloc(1024 * 4);
    size_t o_b1  = alloc(1024 * 4);
    size_t o_lb0 = alloc(1024 * 4);
    size_t o_lb1 = alloc(1024 * 4);
    size_t o_Z0  = alloc((size_t)N0 * 1024 * 2);
    size_t o_Z1  = alloc((size_t)N1 * 1024 * 2);
    size_t o_Y0  = alloc((size_t)N0 * 1024 * 4);
    size_t o_Y1  = alloc((size_t)N1 * 1024 * 4);
    (void)ws_size;

    float* sum0  = (float*)(ws + o_sum0);
    float* cnt0  = (float*)(ws + o_cnt0);
    float* sum1a = (float*)(ws + o_sum1a);
    float* cnt1a = (float*)(ws + o_cnt1a);
    float* sum1c = (float*)(ws + o_sum1c);
    float* cnt1c = (float*)(ws + o_cnt1c);
    unsigned short* A0  = (unsigned short*)(ws + o_A0);
    unsigned short* A1  = (unsigned short*)(ws + o_A1);
    unsigned short* W0T = (unsigned short*)(ws + o_W0T);
    unsigned short* W1T = (unsigned short*)(ws + o_W1T);
    unsigned short* L0T = (unsigned short*)(ws + o_L0T);
    unsigned short* L1T = (unsigned short*)(ws + o_L1T);
    float* bias0 = (float*)(ws + o_b0);
    float* bias1 = (float*)(ws + o_b1);
    float* lb0   = (float*)(ws + o_lb0);
    float* lb1   = (float*)(ws + o_lb1);
    unsigned short* Z0 = (unsigned short*)(ws + o_Z0);
    unsigned short* Z1 = (unsigned short*)(ws + o_Z1);
    float* Y0 = (float*)(ws + o_Y0);
    float* Y1 = (float*)(ws + o_Y1);

    float* out = (float*)d_out;
    float* h0  = out;
    float* h1  = out + (size_t)N0 * HH;
    float* c0n = out + (size_t)(N0 + N1) * HH;
    float* c1n = out + (size_t)(2 * N0 + N1) * HH;
    (void)out_size;

    // 1) zero accumulators
    zero_f32<<<2048, 256, 0, stream>>>((float*)ws, (long long)(zeroEnd / 4));

    // 2) scatter-mean sums (gate-independent -> computed once, not 4x)
    scatter_sum<<<8192, 256, 0, stream>>>(x1, e1, E1, sum0, cnt0);    // n1 -> n0
    scatter_sum<<<8192, 256, 0, stream>>>(x0, e0, E0, sum1a, cnt1a);  // n0 -> n1
    scatter_sum<<<8192, 256, 0, stream>>>(x1, e2, E2, sum1c, cnt1c);  // n1 -> n1

    // 3) pack bf16 activations and weights
    pack_a0<<<4096, 256, 0, stream>>>(sum0, cnt0, x0, A0, N0);
    pack_a1<<<4096, 256, 0, stream>>>(sum1a, cnt1a, sum1c, cnt1c, x1, A1, N1);
    pack_weights<<<1536, 256, 0, stream>>>(P, W0T, W1T, L0T, L1T, bias0, bias1, lb0, lb1);

    // 4) fused SAGE GEMMs: Z = A * Wbig + bias  (bf16 out, all 4 gates stacked)
    int rt0 = N0 / 32, rt1 = N1 / 32;   // 32-row wave tiles
    dim3 g0((rt0 + GEMM_WAVES - 1) / GEMM_WAVES, 16);
    dim3 g1((rt1 + GEMM_WAVES - 1) / GEMM_WAVES, 16);
    gemm_wmma<1><<<g0, GEMM_WAVES * 32, 0, stream>>>(A0, 256, W0T, 256, bias0, Z0, 1024, rt0, 0);
    gemm_wmma<1><<<g1, GEMM_WAVES * 32, 0, stream>>>(A1, 384, W1T, 384, bias1, Z1, 1024, rt1, 0);

    // 5) per-gate 256->256 linears (block-diagonal GEMM, f32 out)
    gemm_wmma<0><<<g0, GEMM_WAVES * 32, 0, stream>>>(Z0, 1024, L0T, 256, lb0, Y0, 1024, rt0, 256);
    gemm_wmma<0><<<g1, GEMM_WAVES * 32, 0, stream>>>(Z1, 1024, L1T, 256, lb1, Y1, 1024, rt1, 256);

    // 6) activations + LSTM combine
    lstm_combine<<<8192, 256, 0, stream>>>(Y0, c0, h0, c0n, N0);
    lstm_combine<<<8192, 256, 0, stream>>>(Y1, c1, h1, c1n, N1);
}
